// Block_81037442941553
// MI455X (gfx1250) — matrix operand order repaired and checked by reading, not compiled, not run on hardware
//
#include <hip/hip_runtime.h>
#include <hip/hip_bf16.h>

// ---------------------------------------------------------------------------
// Transformer block forward for MI455X (gfx1250, wave32, WMMA).
// bf16 compute / f32 accumulate via v_wmma_f32_16x16x32_bf16.
// GEMMs: 128x256 block tile, 64x64 wave tile (16 WMMA / K-step), double-
// buffered LDS, async global->LDS (GLOBAL_LOAD_ASYNC_TO_LDS_B128) for A tiles.
// ---------------------------------------------------------------------------

typedef __attribute__((ext_vector_type(16))) __bf16 v16bf;
typedef __attribute__((ext_vector_type(8)))  float  v8f;
typedef __attribute__((ext_vector_type(4)))  int    v4i_;

#define Bq_ 4
#define Sq_ 2048
#define Eq_ 1024
#define Hq_ 16
#define Dq_ 64
#define FFq_ 6144
#define Mq_ (Bq_*Sq_)          // 8192 rows

#if defined(__gfx1250__) && __has_builtin(__builtin_amdgcn_global_load_async_to_lds_b128)
#define USE_ASYNC 1
#else
#define USE_ASYNC 0
#endif

__device__ __forceinline__ unsigned short f2bf(float f) {
  unsigned int u = __float_as_uint(f);
  unsigned int r = u + 0x7fffu + ((u >> 16) & 1u);   // round-to-nearest-even
  return (unsigned short)(r >> 16);
}

union Frag { v16bf v; unsigned int u[8]; };

#if USE_ASYNC
__device__ __forceinline__ void async_ld_b128(const void* g, void* l) {
  __builtin_amdgcn_global_load_async_to_lds_b128(
      (v4i_ __attribute__((address_space(1)))*)(g),
      (v4i_ __attribute__((address_space(3)))*)(l), 0, 0);
}
__device__ __forceinline__ void wait_async0() {
#if __has_builtin(__builtin_amdgcn_s_wait_asynccnt)
  __builtin_amdgcn_s_wait_asynccnt(0);
#else
  asm volatile("s_wait_asynccnt 0" ::: "memory");
#endif
}
#endif

// ---------------------------------------------------------------------------
// Weight conversion kernels (run every launch; deterministic)
// ---------------------------------------------------------------------------
__global__ __launch_bounds__(256) void conv_f32_bf16(const float* __restrict__ in,
                                                     unsigned short* __restrict__ out,
                                                     size_t n) {
  size_t i = (size_t)blockIdx.x * 256 + threadIdx.x;
  if (i < n) out[i] = f2bf(in[i]);
}

// Wq/Wk/Wv [H,E,D] f32  ->  packed bf16 [E, 3*E], col = which*1024 + h*64 + d
__global__ __launch_bounds__(256) void conv_qkvw(const float* __restrict__ Wq,
                                                 const float* __restrict__ Wk,
                                                 const float* __restrict__ Wv,
                                                 unsigned short* __restrict__ out) {
  size_t i = (size_t)blockIdx.x * 256 + threadIdx.x;   // 3*2^20 total
  if (i >= (size_t)3 * Hq_ * Eq_ * Dq_) return;
  int d = (int)(i & 63);
  int e = (int)((i >> 6) & 1023);
  int h = (int)((i >> 16) & 15);
  int which = (int)(i >> 20);
  const float* W = (which == 0) ? Wq : (which == 1) ? Wk : Wv;
  out[(size_t)e * (3 * Eq_) + which * Eq_ + h * Dq_ + d] =
      f2bf(W[((size_t)h * Eq_ + e) * Dq_ + d]);
}

__global__ __launch_bounds__(256) void conv_qkvb(const float* __restrict__ bq,
                                                 const float* __restrict__ bk,
                                                 const float* __restrict__ bv,
                                                 float* __restrict__ out) {
  int i = blockIdx.x * 256 + threadIdx.x;
  if (i >= 3 * Eq_) return;
  int which = i >> 10;
  const float* bb = (which == 0) ? bq : (which == 1) ? bk : bv;
  out[i] = bb[i & 1023];
}

// ---------------------------------------------------------------------------
// LayerNorm: one block per row of [M, 1024], writes bf16
// ---------------------------------------------------------------------------
__global__ __launch_bounds__(256) void ln_kernel(const float* __restrict__ x,
                                                 const float* __restrict__ g,
                                                 const float* __restrict__ bb,
                                                 unsigned short* __restrict__ out) {
  const int row = blockIdx.x;
  const float* xr = x + (size_t)row * Eq_;
  __shared__ float red[8];
  float v[4];
  float s = 0.f;
#pragma unroll
  for (int i = 0; i < 4; i++) { v[i] = xr[threadIdx.x + i * 256]; s += v[i]; }
#pragma unroll
  for (int off = 16; off >= 1; off >>= 1) s += __shfl_xor(s, off, 32);
  if ((threadIdx.x & 31) == 0) red[threadIdx.x >> 5] = s;
  __syncthreads();
  float mean = 0.f;
#pragma unroll
  for (int i = 0; i < 8; i++) mean += red[i];
  mean *= (1.f / (float)Eq_);
  float s2 = 0.f;
#pragma unroll
  for (int i = 0; i < 4; i++) { float d = v[i] - mean; s2 += d * d; }
#pragma unroll
  for (int off = 16; off >= 1; off >>= 1) s2 += __shfl_xor(s2, off, 32);
  __syncthreads();
  if ((threadIdx.x & 31) == 0) red[threadIdx.x >> 5] = s2;
  __syncthreads();
  float var = 0.f;
#pragma unroll
  for (int i = 0; i < 8; i++) var += red[i];
  var *= (1.f / (float)Eq_);
  float rstd = rsqrtf(var + 1e-5f);
#pragma unroll
  for (int i = 0; i < 4; i++) {
    int c = threadIdx.x + i * 256;
    out[(size_t)row * Eq_ + c] = f2bf((v[i] - mean) * rstd * g[c] + bb[c]);
  }
}

// ---------------------------------------------------------------------------
// Tiled bf16 WMMA GEMM: C[M,N] = A[M,K] (bf16) x B[K,N] (bf16) + bias
// EPI bit0: add f32 residual [M,N]; bit1: exact GELU; bit2: output bf16
// BM=128 BN=256 BK=32; 8 waves (2x4), wave tile 64x64 (4x4 frags, 16 WMMA).
// Double-buffered LDS, software pipelined; A tile via async global->LDS.
// ---------------------------------------------------------------------------
#define GBM 128
#define GBN 256
#define GBK 32
#define SAKA 40   // A LDS K-stride (bf16): 80B rows -> 16B aligned, conflict-free
#define SAKB 34   // B LDS K-stride (bf16)

template <int EPI>
__global__ __launch_bounds__(256) void gemm_bf16(const unsigned short* __restrict__ A,
                                                 const unsigned short* __restrict__ Bw,
                                                 const float* __restrict__ bias,
                                                 const float* __restrict__ resid,
                                                 void* __restrict__ out,
                                                 int M, int N, int K) {
  __shared__ unsigned short sA[2][GBM * SAKA];
  __shared__ unsigned short sB[2][GBN * SAKB];   // transposed: [n][k]
  const int tid = threadIdx.x;
  const int lane = tid & 31;
  const int wave = tid >> 5;
  const int waveM = wave >> 2;   // 0..1  -> 64-row slab
  const int waveN = wave & 3;    // 0..3  -> 64-col slab
  const int lh = lane >> 4;      // lane half
  const int ln = lane & 15;
  const int m0 = blockIdx.y * GBM;
  const int n0 = blockIdx.x * GBN;

  v8f acc[4][4];
#pragma unroll
  for (int i = 0; i < 4; i++)
#pragma unroll
    for (int j = 0; j < 4; j++) acc[i][j] = (v8f){0.f,0.f,0.f,0.f,0.f,0.f,0.f,0.f};

  const int aRow = tid >> 1;            // 0..127
  const int aCol = (tid & 1) * 16;      // 0 / 16
  const int bRow = tid >> 3;            // 0..31
  const int bCol = (tid & 7) * 32;      // 0..224

  const unsigned short* gA = A + (size_t)(m0 + aRow) * K + aCol;
  const unsigned short* gB = Bw + (size_t)bRow * N + n0 + bCol;
  const int nTiles = K / GBK;

  uint4 bReg[4];
#if !USE_ASYNC
  uint4 aReg[2];
#endif

  // prologue: fetch tile 0
#if USE_ASYNC
  async_ld_b128(gA,     &sA[0][aRow * SAKA + aCol]);
  async_ld_b128(gA + 8, &sA[0][aRow * SAKA + aCol + 8]);
#else
  aReg[0] = ((const uint4*)gA)[0];
  aReg[1] = ((const uint4*)gA)[1];
#endif
#pragma unroll
  for (int i = 0; i < 4; i++) bReg[i] = ((const uint4*)gB)[i];

  int s = 0;
  for (int it = 0; it < nTiles; ++it) {
    // store staged B (transposed) [+ A in fallback mode] into stage s
    {
      alignas(16) unsigned short tmp[32];
#pragma unroll
      for (int i = 0; i < 4; i++) ((uint4*)tmp)[i] = bReg[i];
#pragma unroll
      for (int i = 0; i < 32; i++) sB[s][(bCol + i) * SAKB + bRow] = tmp[i];
    }
#if !USE_ASYNC
    {
      unsigned int* sp = (unsigned int*)(&sA[s][aRow * SAKA + aCol]);
      sp[0] = aReg[0].x; sp[1] = aReg[0].y; sp[2] = aReg[0].z; sp[3] = aReg[0].w;
      sp[4] = aReg[1].x; sp[5] = aReg[1].y; sp[6] = aReg[1].z; sp[7] = aReg[1].w;
    }
#endif
#if USE_ASYNC
    wait_async0();           // stage-s A tile landed in LDS
#endif
    __syncthreads();

    // issue next tile's loads (overlap with compute below)
    if (it + 1 < nTiles) {
      const unsigned short* gA2 = gA + (it + 1) * GBK;
      const unsigned short* gB2 = gB + (size_t)(it + 1) * GBK * N;
#if USE_ASYNC
      async_ld_b128(gA2,     &sA[s ^ 1][aRow * SAKA + aCol]);
      async_ld_b128(gA2 + 8, &sA[s ^ 1][aRow * SAKA + aCol + 8]);
#else
      aReg[0] = ((const uint4*)gA2)[0];
      aReg[1] = ((const uint4*)gA2)[1];
#endif
#pragma unroll
      for (int i = 0; i < 4; i++) bReg[i] = ((const uint4*)gB2)[i];
    }

    // compute from stage s: 16 WMMAs
    Frag af[4];
#pragma unroll
    for (int mf = 0; mf < 4; mf++) {
      const unsigned short* base = sA[s] + (waveM * 64 + mf * 16 + ln) * SAKA;
#pragma unroll
      for (int j = 0; j < 8; j++) {
        int kk = (lh << 4) + (j << 1);
        af[mf].u[j] = *(const unsigned int*)(base + kk);
      }
    }
#pragma unroll
    for (int nf = 0; nf < 4; nf++) {
      Frag bfr;
      const unsigned short* base = sB[s] + (waveN * 64 + nf * 16 + ln) * SAKB + (lh << 4);
#pragma unroll
      for (int j = 0; j < 8; j++)
        bfr.u[j] = *(const unsigned int*)(base + (j << 1));
#pragma unroll
      for (int mf = 0; mf < 4; mf++)
        acc[mf][nf] = __builtin_amdgcn_wmma_f32_16x16x32_bf16(
            false, af[mf].v, false, bfr.v, (short)0, acc[mf][nf], false, false);
    }
    s ^= 1;
  }

  // epilogue: D layout m = r + 8*lh, n = ln
#pragma unroll
  for (int mf = 0; mf < 4; mf++) {
#pragma unroll
    for (int nf = 0; nf < 4; nf++) {
      int col = n0 + waveN * 64 + nf * 16 + ln;
      float bv = bias[col];
#pragma unroll
      for (int r = 0; r < 8; r++) {
        int row = m0 + waveM * 64 + mf * 16 + r + (lh << 3);
        float v = acc[mf][nf][r] + bv;
        if (EPI & 1) v += resid[(size_t)row * N + col];
        if (EPI & 2) v = 0.5f * v * (1.0f + erff(v * 0.70710678118654752f));
        if (EPI & 4) ((unsigned short*)out)[(size_t)row * N + col] = f2bf(v);
        else         ((float*)out)[(size_t)row * N + col] = v;
      }
    }
  }
}

// ---------------------------------------------------------------------------
// Flash attention (causal). qkv bf16 [B*S, 3072]: cols [0,1024)=Q [1024,2048)=K
// [2048,3072)=V (per-head 64 contiguous). Output o bf16 [B*S, 1024] head-concat.
// Block = 8 waves, 128 Q rows per block (16 per wave); key tiles of 64.
// ---------------------------------------------------------------------------
#define SKS 66   // padded stride for 64-wide tiles

__global__ __launch_bounds__(256) void attn_kernel(const unsigned short* __restrict__ qkv,
                                                   unsigned short* __restrict__ o) {
  const int bh = blockIdx.y;
  const int b = bh / Hq_;
  const int h = bh % Hq_;
  const int q0 = blockIdx.x * 128;
  const int tid = threadIdx.x, lane = tid & 31, wave = tid >> 5;
  const int lh = lane >> 4, ln = lane & 15;

  __shared__ unsigned short sK[64 * SKS];        // [t][d]
  __shared__ unsigned short sV[64 * SKS];        // transposed [d][t]
  __shared__ unsigned short sP[8 * 16 * SKS];    // wave-private P tiles [m][t]

  const size_t rs = 3 * Eq_;
  const unsigned short* qbase = qkv + (size_t)b * Sq_ * rs + h * Dq_;
  const unsigned short* kbase = qbase + Eq_;
  const unsigned short* vbase = qbase + 2 * Eq_;

  // Q fragments (A layout), rows q0 + wave*16 + m
  Frag qa[2];
  {
    int s = q0 + wave * 16 + ln;
    const unsigned short* qr = qbase + (size_t)s * rs;
#pragma unroll
    for (int kf = 0; kf < 2; kf++)
#pragma unroll
      for (int j = 0; j < 8; j++) {
        int d = kf * 32 + (lh << 4) + (j << 1);
        qa[kf].u[j] = *(const unsigned int*)(qr + d);
      }
  }

  v8f oc[4];
#pragma unroll
  for (int i = 0; i < 4; i++) oc[i] = (v8f){0.f,0.f,0.f,0.f,0.f,0.f,0.f,0.f};
  float mrow[8], lrow[8];
#pragma unroll
  for (int r = 0; r < 8; r++) { mrow[r] = -3.0e38f; lrow[r] = 0.f; }

  const int srow_base = q0 + wave * 16 + (lh << 3);

  for (int t0 = 0; t0 < q0 + 128; t0 += 64) {
    __syncthreads();
    { // stage K [t][d] and V transposed [d][t]
      int tr = tid >> 2;
      int dc = (tid & 3) * 16;
      const uint4* gk = (const uint4*)(kbase + (size_t)(t0 + tr) * rs + dc);
      uint4 d0 = gk[0], d1 = gk[1];
      unsigned int* sp = (unsigned int*)(sK + tr * SKS + dc);
      sp[0] = d0.x; sp[1] = d0.y; sp[2] = d0.z; sp[3] = d0.w;
      sp[4] = d1.x; sp[5] = d1.y; sp[6] = d1.z; sp[7] = d1.w;
      const uint4* gv = (const uint4*)(vbase + (size_t)(t0 + tr) * rs + dc);
      uint4 e0 = gv[0], e1 = gv[1];
      alignas(16) unsigned short tmp[16];
      ((uint4*)tmp)[0] = e0; ((uint4*)tmp)[1] = e1;
#pragma unroll
      for (int i = 0; i < 16; i++) sV[(dc + i) * SKS + tr] = tmp[i];
    }
    __syncthreads();

    // scores: S = Q (16x64) * K^T (64x64)  -> 4 n-frags, 2 k-steps each
    v8f sc[4];
#pragma unroll
    for (int nf = 0; nf < 4; nf++) {
      v8f c = (v8f){0.f,0.f,0.f,0.f,0.f,0.f,0.f,0.f};
#pragma unroll
      for (int kf = 0; kf < 2; kf++) {
        Frag kb;
        const unsigned short* base = sK + (nf * 16 + ln) * SKS + kf * 32 + (lh << 4);
#pragma unroll
        for (int j = 0; j < 8; j++) kb.u[j] = *(const unsigned int*)(base + (j << 1));
        c = __builtin_amdgcn_wmma_f32_16x16x32_bf16(false, qa[kf].v, false, kb.v,
                                                    (short)0, c, false, false);
      }
      sc[nf] = c;
    }

    // scale + causal mask + tile row-max
    float tmax[8];
#pragma unroll
    for (int r = 0; r < 8; r++) tmax[r] = -3.0e38f;
#pragma unroll
    for (int nf = 0; nf < 4; nf++) {
      int t = t0 + nf * 16 + ln;
#pragma unroll
      for (int r = 0; r < 8; r++) {
        float v = sc[nf][r] * 0.125f;
        if (t > srow_base + r) v = -3.0e38f;
        sc[nf][r] = v;
        tmax[r] = fmaxf(tmax[r], v);
      }
    }
#pragma unroll
    for (int r = 0; r < 8; r++) {
      float v = tmax[r];
      v = fmaxf(v, __shfl_xor(v, 1, 32));
      v = fmaxf(v, __shfl_xor(v, 2, 32));
      v = fmaxf(v, __shfl_xor(v, 4, 32));
      v = fmaxf(v, __shfl_xor(v, 8, 32));
      tmax[r] = v;
    }
    float fac[8];
#pragma unroll
    for (int r = 0; r < 8; r++) {
      float nm = fmaxf(mrow[r], tmax[r]);
      fac[r] = __expf(mrow[r] - nm);
      mrow[r] = nm;
    }
    // P = exp(S - m), row sums, stage bf16 P into wave-private LDS
    float tsum[8];
#pragma unroll
    for (int r = 0; r < 8; r++) tsum[r] = 0.f;
#pragma unroll
    for (int nf = 0; nf < 4; nf++) {
#pragma unroll
      for (int r = 0; r < 8; r++) {
        float p = __expf(sc[nf][r] - mrow[r]);
        tsum[r] += p;
        sP[(wave * 16 + r + (lh << 3)) * SKS + nf * 16 + ln] = f2bf(p);
      }
    }
#pragma unroll
    for (int r = 0; r < 8; r++) {
      float v = tsum[r];
      v += __shfl_xor(v, 1, 32);
      v += __shfl_xor(v, 2, 32);
      v += __shfl_xor(v, 4, 32);
      v += __shfl_xor(v, 8, 32);
      lrow[r] = lrow[r] * fac[r] + v;
    }
#pragma unroll
    for (int nf = 0; nf < 4; nf++)
#pragma unroll
      for (int r = 0; r < 8; r++) oc[nf][r] *= fac[r];

    // O += P (16x64) * V (64x64): reload P as A-frags from LDS
    Frag pa[2];
    {
      const unsigned short* pr = sP + (wave * 16 + ln) * SKS;
#pragma unroll
      for (int kf = 0; kf < 2; kf++)
#pragma unroll
        for (int j = 0; j < 8; j++) {
          int kk = kf * 32 + (lh << 4) + (j << 1);
          pa[kf].u[j] = *(const unsigned int*)(pr + kk);
        }
    }
#pragma unroll
    for (int nf = 0; nf < 4; nf++) {
#pragma unroll
      for (int kf = 0; kf < 2; kf++) {
        Frag vb;
        const unsigned short* base = sV + (nf * 16 + ln) * SKS + kf * 32 + (lh << 4);
#pragma unroll
        for (int j = 0; j < 8; j++) vb.u[j] = *(const unsigned int*)(base + (j << 1));
        oc[nf] = __builtin_amdgcn_wmma_f32_16x16x32_bf16(false, pa[kf].v, false, vb.v,
                                                         (short)0, oc[nf], false, false);
      }
    }
  }

  // normalize and store O (head-concat layout)
#pragma unroll
  for (int r = 0; r < 8; r++) lrow[r] = 1.f / lrow[r];
#pragma unroll
  for (int nf = 0; nf < 4; nf++) {
#pragma unroll
    for (int r = 0; r < 8; r++) {
      int s = q0 + wave * 16 + r + (lh << 3);
      int d = nf * 16 + ln;
      o[(size_t)(b * Sq_ + s) * Eq_ + h * Dq_ + d] = f2bf(oc[nf][r] * lrow[r]);
    }
  }
}

// ---------------------------------------------------------------------------
// Launch
// ---------------------------------------------------------------------------
extern "C" void kernel_launch(void* const* d_in, const int* in_sizes, int n_in,
                              void* d_out, int out_size, void* d_ws, size_t ws_size,
                              hipStream_t stream) {
  const float* x     = (const float*)d_in[0];
  const float* ln1_g = (const float*)d_in[1];
  const float* ln1_b = (const float*)d_in[2];
  const float* Wq    = (const float*)d_in[3];
  const float* bq    = (const float*)d_in[4];
  const float* Wk    = (const float*)d_in[5];
  const float* bk    = (const float*)d_in[6];
  const float* Wv    = (const float*)d_in[7];
  const float* bv    = (const float*)d_in[8];
  const float* Wc    = (const float*)d_in[9];
  const float* bc    = (const float*)d_in[10];
  const float* ln2_g = (const float*)d_in[11];
  const float* ln2_b = (const float*)d_in[12];
  const float* W1    = (const float*)d_in[13];
  const float* b1    = (const float*)d_in[14];
  const float* W2    = (const float*)d_in[15];
  const float* b2    = (const float*)d_in[16];

  char* w = (char*)d_ws;
  size_t off = 0;
  auto alloc = [&](size_t bytes) { void* p = w + off; off = (off + bytes + 255) & ~(size_t)255; return p; };

  unsigned short* qkvw = (unsigned short*)alloc((size_t)Eq_ * 3 * Eq_ * 2);
  unsigned short* wcb  = (unsigned short*)alloc((size_t)Eq_ * Eq_ * 2);
  unsigned short* w1b  = (unsigned short*)alloc((size_t)Eq_ * FFq_ * 2);
  unsigned short* w2b  = (unsigned short*)alloc((size_t)FFq_ * Eq_ * 2);
  float*          bqkv = (float*)alloc((size_t)3 * Eq_ * 4);
  unsigned short* xn   = (unsigned short*)alloc((size_t)Mq_ * Eq_ * 2);
  unsigned short* qkvb = (unsigned short*)alloc((size_t)Mq_ * 3 * Eq_ * 2);
  unsigned short* obuf = (unsigned short*)alloc((size_t)Mq_ * Eq_ * 2);
  float*          x1   = (float*)alloc((size_t)Mq_ * Eq_ * 4);
  unsigned short* xn2  = (unsigned short*)alloc((size_t)Mq_ * Eq_ * 2);
  unsigned short* hbuf = (unsigned short*)alloc((size_t)Mq_ * FFq_ * 2);
  (void)ws_size; (void)n_in; (void)in_sizes; (void)out_size;

  // 1) weight/bias conversion
  conv_qkvw<<<(3u << 20) / 256, 256, 0, stream>>>(Wq, Wk, Wv, qkvw);
  conv_qkvb<<<12, 256, 0, stream>>>(bq, bk, bv, bqkv);
  conv_f32_bf16<<<((size_t)Eq_ * Eq_ + 255) / 256, 256, 0, stream>>>(Wc, wcb, (size_t)Eq_ * Eq_);
  conv_f32_bf16<<<((size_t)Eq_ * FFq_ + 255) / 256, 256, 0, stream>>>(W1, w1b, (size_t)Eq_ * FFq_);
  conv_f32_bf16<<<((size_t)FFq_ * Eq_ + 255) / 256, 256, 0, stream>>>(W2, w2b, (size_t)FFq_ * Eq_);

  // 2) LN1
  ln_kernel<<<Mq_, 256, 0, stream>>>(x, ln1_g, ln1_b, xn);

  // 3) QKV projection: [8192,1024] x [1024,3072] -> bf16
  gemm_bf16<4><<<dim3(3 * Eq_ / GBN, Mq_ / GBM), 256, 0, stream>>>(
      xn, qkvw, bqkv, nullptr, qkvb, Mq_, 3 * Eq_, Eq_);

  // 4) causal flash attention -> obuf (bf16, head-concat)
  attn_kernel<<<dim3(Sq_ / 128, Bq_ * Hq_), 256, 0, stream>>>(qkvb, obuf);

  // 5) output projection + residual -> x1 (f32)
  gemm_bf16<1><<<dim3(Eq_ / GBN, Mq_ / GBM), 256, 0, stream>>>(
      obuf, wcb, bc, x, x1, Mq_, Eq_, Eq_);

  // 6) LN2
  ln_kernel<<<Mq_, 256, 0, stream>>>(x1, ln2_g, ln2_b, xn2);

  // 7) MLP up + exact GELU -> bf16
  gemm_bf16<6><<<dim3(FFq_ / GBN, Mq_ / GBM), 256, 0, stream>>>(
      xn2, w1b, b1, nullptr, hbuf, Mq_, FFq_, Eq_);

  // 8) MLP down + residual -> d_out (f32)
  gemm_bf16<1><<<dim3(Eq_ / GBN, Mq_ / GBM), 256, 0, stream>>>(
      hbuf, w2b, b2, x1, (float*)d_out, Mq_, Eq_, FFq_);
}
